// PointCLIP_Model_87943750353319
// MI455X (gfx1250) — compile-verified, matrix operand order checked by reading, not run on hardware
//
#include <hip/hip_runtime.h>
#include <hip/hip_bf16.h>
#include <math.h>

// ---------------------------------------------------------------------------
// PointCLIP adapter + Sinkhorn logits for MI455X (gfx1250, wave32, WMMA).
// GEMMs: bf16 operands; B tiles DMA'd to LDS via the Tensor Data Mover
// (tensor_load_to_lds, double-buffered, TENSORcnt), A via global_load_b128,
// compute via v_wmma_f32_16x16x32_bf16.
// ---------------------------------------------------------------------------

#define BB 512
#define VV 10
#define DD 1024
#define CC 100
#define PP 4

typedef __attribute__((ext_vector_type(16))) __bf16 v16bf;
typedef __attribute__((ext_vector_type(8)))  float  v8f;
typedef __attribute__((ext_vector_type(4))) unsigned tdm_g0_t;
typedef __attribute__((ext_vector_type(8))) int      tdm_g1_t;
typedef __attribute__((ext_vector_type(4))) int      tdm_g2_t;
typedef __attribute__((ext_vector_type(8))) int      tdm_g4_t;

union FragB { uint4 q[2]; v16bf v; };
union Pack8 { __bf16 h[8]; uint4 q; };

// ---- Tensor Data Mover: DMA a (rows x 32) bf16 tile (row-major, leading dim
// ldK elements) from global memory into LDS at byte offset lds_off.
// D# packing per CDNA5 ISA:
//   group0: count=1 | lds_addr | global_addr | type=2
//   group1: data_size=2B, tensor_dim0=tile_dim0=32, tensor_dim1=tile_dim1=rows,
//           tensor_dim0_stride=ldK  (2-D tile -> remaining groups zero)
__device__ __forceinline__ void tdm_load_tile(const __bf16* gsrc, int ldK,
                                              unsigned lds_off, int rows) {
  unsigned long long ga = (unsigned long long)(uintptr_t)gsrc;
  tdm_g0_t g0;
  g0[0] = 1u;                                          // count=1, user mode
  g0[1] = lds_off;                                     // LDS byte address
  g0[2] = (unsigned)(ga & 0xFFFFFFFFull);              // global_addr[31:0]
  g0[3] = (unsigned)((ga >> 32) & 0x1FFFFFFull)        // global_addr[56:32]
          | (2u << 30);                                 // type = 2 ("image")
  tdm_g1_t g1;
  g1[0] = (int)(1u << 16);                             // data_size: 1 -> 2 bytes
  g1[1] = (int)(32u << 16);                            // tensor_dim0[15:0] = 32
  g1[2] = (int)(((unsigned)rows & 0xFFFFu) << 16);     // tensor_dim1[15:0] = rows
  g1[3] = (int)(32u << 16);                            // tile_dim0 = 32
  g1[4] = (int)((unsigned)rows & 0xFFFFu);             // tile_dim1 = rows
  g1[5] = (int)(unsigned)ldK;                          // tensor_dim0_stride[31:0]
  g1[6] = 0;                                           // stride hi / dim1_stride
  g1[7] = 0;
  tdm_g2_t gz4 = {0, 0, 0, 0};
  tdm_g4_t gz8 = {0, 0, 0, 0, 0, 0, 0, 0};
  __builtin_amdgcn_tensor_load_to_lds(g0, g1, gz4, gz4, gz8, 0);
}

// Load one 16x32 bf16 A fragment (ISA 7.12.2 layout) from row-major global bf16.
__device__ __forceinline__ v16bf load_frag_g(const __bf16* __restrict__ base, int ld,
                                             int row0, int k0, int lane) {
  int r = lane & 15;
  int h = lane >> 4;
  const __bf16* p0 = base + (size_t)(row0 + r) * ld + k0 + h * 8;
  FragB f;
  f.q[0] = *(const uint4*)p0;
  f.q[1] = *(const uint4*)(p0 + 16);
  return f.v;
}

// Load one 16x32 bf16 B fragment from an LDS-resident (rows x 32) tile.
__device__ __forceinline__ v16bf load_frag_lds(const __bf16* tile, int nlocal, int lane) {
  int r = nlocal + (lane & 15);
  int h = lane >> 4;
  const __bf16* p0 = tile + r * 32 + h * 8;
  FragB f;
  f.q[0] = *(const uint4*)p0;
  f.q[1] = *(const uint4*)(p0 + 16);
  return f.v;
}

// C(M,N) = epi( A(M,K) * B(N,K)^T + bias ).  A, B row-major bf16 (torch weight
// layout for B).  Block: 8 waves stacked in M, each wave 32 x (NT*16).
// B tile (NT*16 x 32) is TDM-DMA'd into LDS once per block per K-step,
// double-buffered; waves consume via ds_load_b128.
// EPI: 0=none, 1=+bias, 2=+bias,relu, 3=+bias,relu, *0.6 + res*0.4
template <int NT, int EPI, bool OUT_BF16>
__global__ __launch_bounds__(256) void gemm_bf16_wmma(
    const __bf16* __restrict__ A, int lda,
    const __bf16* __restrict__ B, int ldb,
    void* __restrict__ Cout, int ldc,
    const float* __restrict__ bias,
    const float* __restrict__ res,
    int K) {
  constexpr int MT = 2;
  constexpr int BN = NT * 16;
  __shared__ __align__(16) __bf16 ldsB[2][BN * 32];

  int lane = threadIdx.x & 31;
  int wave = threadIdx.x >> 5;
  int mr = (blockIdx.y * 8 + wave) * (16 * MT);
  int n0 = blockIdx.x * BN;

  const __bf16* bsrc = B + (size_t)n0 * ldb;
  unsigned lds_off0 = (unsigned)(uintptr_t)&ldsB[0][0];
  unsigned lds_off1 = (unsigned)(uintptr_t)&ldsB[1][0];

  // Prologue: DMA first B tile, wait, share with all waves.
  if (wave == 0) {
    tdm_load_tile(bsrc, ldb, lds_off0, BN);
    __builtin_amdgcn_s_wait_tensorcnt(0);
  }
  __syncthreads();

  v8f acc[MT][NT];
  v8f zero = {0.f, 0.f, 0.f, 0.f, 0.f, 0.f, 0.f, 0.f};
#pragma unroll
  for (int i = 0; i < MT; ++i)
#pragma unroll
    for (int j = 0; j < NT; ++j) acc[i][j] = zero;

  int nk = K / 32;
  for (int kk = 0; kk < nk; ++kk) {
    int cur = kk & 1;
    // Kick off DMA of the next B tile into the other buffer (overlaps compute).
    if (wave == 0 && (kk + 1) < nk)
      tdm_load_tile(bsrc + (size_t)(kk + 1) * 32, ldb,
                    cur ? lds_off0 : lds_off1, BN);

    int k0 = kk * 32;
    v16bf a[MT];
#pragma unroll
    for (int i = 0; i < MT; ++i) a[i] = load_frag_g(A, lda, mr + i * 16, k0, lane);
#pragma unroll
    for (int j = 0; j < NT; ++j) {
      v16bf b = load_frag_lds(&ldsB[cur][0], j * 16, lane);
#pragma unroll
      for (int i = 0; i < MT; ++i) {
        acc[i][j] = __builtin_amdgcn_wmma_f32_16x16x32_bf16(
            false, a[i], false, b, (short)0, acc[i][j], false, false);
      }
    }

    if (wave == 0) __builtin_amdgcn_s_wait_tensorcnt(0);
    __syncthreads();
  }

  // C/D layout: lane (0..15) -> N = n + lane, reg e -> M = base + e (+8 hi half)
  int nl = lane & 15;
  int hh = lane >> 4;
#pragma unroll
  for (int i = 0; i < MT; ++i) {
#pragma unroll
    for (int j = 0; j < NT; ++j) {
      int ncol = n0 + j * 16 + nl;
#pragma unroll
      for (int e = 0; e < 8; ++e) {
        int mrow = mr + i * 16 + hh * 8 + e;
        float val = acc[i][j][e];
        if (EPI >= 1) val += bias[ncol];
        if (EPI >= 2) val = fmaxf(val, 0.f);
        if (EPI == 3) val = val * 0.6f + res[(size_t)mrow * ldc + ncol] * 0.4f;
        if (OUT_BF16)
          ((__bf16*)Cout)[(size_t)mrow * ldc + ncol] = (__bf16)val;
        else
          ((float*)Cout)[(size_t)mrow * ldc + ncol] = val;
      }
    }
  }
}

// Bulk f32 -> bf16 conversion, 8 elements/thread, 16B stores.
__global__ __launch_bounds__(256) void cvt_bf16_kernel(
    const float* __restrict__ in, __bf16* __restrict__ out, int n) {
  int i = (blockIdx.x * 256 + threadIdx.x) * 8;
  if (i >= n) return;
  float4 a = *(const float4*)(in + i);
  float4 b = *(const float4*)(in + i + 4);
  Pack8 p;
  p.h[0] = (__bf16)a.x; p.h[1] = (__bf16)a.y;
  p.h[2] = (__bf16)a.z; p.h[3] = (__bf16)a.w;
  p.h[4] = (__bf16)b.x; p.h[5] = (__bf16)b.y;
  p.h[6] = (__bf16)b.z; p.h[7] = (__bf16)b.w;
  *(uint4*)(out + i) = p.q;
}

// Per-column batch-norm statistics (biased variance), folded to scale/shift.
__global__ __launch_bounds__(256) void bn_stats_kernel(
    const float* __restrict__ x, const float* __restrict__ fr,
    const float* __restrict__ gamma, const float* __restrict__ beta,
    float* __restrict__ scale, float* __restrict__ shift,
    int rows, int useFusion) {
  int d = blockIdx.x;
  int t = threadIdx.x;
  float s = 0.f, s2 = 0.f;
  for (int r = t; r < rows; r += 256) {
    float v = x[(size_t)r * DD + d];
    if (useFusion) v *= fr[r % VV];
    s += v; s2 += v * v;
  }
  __shared__ float ls[256], ls2[256];
  ls[t] = s; ls2[t] = s2;
  __syncthreads();
  for (int off = 128; off > 0; off >>= 1) {
    if (t < off) { ls[t] += ls[t + off]; ls2[t] += ls2[t + off]; }
    __syncthreads();
  }
  if (t == 0) {
    float m = ls[0] / rows;
    float var = ls2[0] / rows - m * m;
    float rstd = rsqrtf(var + 1e-5f);
    float sc = rstd * gamma[d];
    scale[d] = sc;
    shift[d] = beta[d] - m * sc;
  }
}

// Elementwise BN apply (+optional fusion, +optional ReLU), bf16 output.
__global__ __launch_bounds__(256) void bn_apply_kernel(
    const float* __restrict__ in, const float* __restrict__ fr,
    const float* __restrict__ scale, const float* __restrict__ shift,
    __bf16* __restrict__ out, int n, int useFusion, int doRelu) {
  int i = blockIdx.x * 256 + threadIdx.x;
  if (i >= n) return;
  int d = i & (DD - 1);
  float v = in[i];
  if (useFusion) v *= fr[(i >> 10) % VV];
  v = v * scale[d] + shift[d];
  if (doRelu) v = fmaxf(v, 0.f);
  out[i] = (__bf16)v;
}

// Row L2-normalize (width DD), f32 in -> bf16 out. One block per row.
__global__ __launch_bounds__(256) void row_norm_kernel(
    const float* __restrict__ in, __bf16* __restrict__ out) {
  int row = blockIdx.x;
  int t = threadIdx.x;
  const float* p = in + (size_t)row * DD;
  float s = 0.f;
  for (int d = t; d < DD; d += 256) { float v = p[d]; s += v * v; }
  __shared__ float ls[256];
  ls[t] = s;
  __syncthreads();
  for (int off = 128; off > 0; off >>= 1) {
    if (t < off) ls[t] += ls[t + off];
    __syncthreads();
  }
  float inv = 1.0f / sqrtf(ls[0]);
  for (int d = t; d < DD; d += 256)
    out[(size_t)row * DD + d] = (__bf16)(p[d] * inv);
}

// One thread per (b, c) Sinkhorn problem (V=10 x P=4), register-resident.
__global__ __launch_bounds__(256) void sinkhorn_logits_kernel(
    const float* __restrict__ sim, const float* __restrict__ logit_scale,
    float* __restrict__ out) {
  int n = blockIdx.x * 256 + threadIdx.x;
  if (n >= BB * CC) return;
  int b = n / CC;
  int cl = n % CC;

  float wd[VV][PP], Ke[VV][PP];
#pragma unroll
  for (int v = 0; v < VV; ++v) {
    const float4 s4 = *(const float4*)(sim + (size_t)(b * VV + v) * (CC * PP) + cl * PP);
    float sv[4] = {s4.x, s4.y, s4.z, s4.w};
#pragma unroll
    for (int p = 0; p < PP; ++p) {
      float w = 1.0f - sv[p];
      wd[v][p] = w;
      Ke[v][p] = expf(-w * 100.0f);   // exp(-wdist / 0.01)
    }
  }

  float r[VV], c[PP];
#pragma unroll
  for (int v = 0; v < VV; ++v) r[v] = 1.0f;
#pragma unroll
  for (int p = 0; p < PP; ++p) c[p] = 1.0f;

  float err = 3.402823e38f;
  int it = 0;
  while (it < 1000 && err > 1e-2f) {
    err = 0.f;
#pragma unroll
    for (int v = 0; v < VV; ++v) {
      float s = 0.f;
#pragma unroll
      for (int p = 0; p < PP; ++p) s += Ke[v][p] * c[p];
      float rn = (1.0f / VV) / s;
      err += fabsf(rn - r[v]);
      r[v] = rn;
    }
#pragma unroll
    for (int p = 0; p < PP; ++p) {
      float s = 0.f;
#pragma unroll
      for (int v = 0; v < VV; ++v) s += Ke[v][p] * r[v];
      c[p] = (1.0f / PP) / s;
    }
    err *= (1.0f / VV);
    ++it;
  }

  float dot = 0.f;
#pragma unroll
  for (int v = 0; v < VV; ++v)
#pragma unroll
    for (int p = 0; p < PP; ++p)
      dot += r[v] * Ke[v][p] * c[p] * wd[v][p];

  out[(size_t)b * CC + cl] = expf(logit_scale[0]) * (1.0f - dot);
}

extern "C" void kernel_launch(void* const* d_in, const int* in_sizes, int n_in,
                              void* d_out, int out_size, void* d_ws, size_t ws_size,
                              hipStream_t stream) {
  const float* feat = (const float*)d_in[0];   // (5120, 1024)
  const float* text = (const float*)d_in[1];   // (100, 4, 1024)
  const float* fr   = (const float*)d_in[2];   // (10,)
  const float* bn1g = (const float*)d_in[3];
  const float* bn1b = (const float*)d_in[4];
  const float* Wg   = (const float*)d_in[5];   // (1024, 10240)
  const float* bg   = (const float*)d_in[6];
  const float* bn2g = (const float*)d_in[7];
  const float* bn2b = (const float*)d_in[8];
  const float* Wv1  = (const float*)d_in[9];   // (1024, 1024)
  const float* bv1  = (const float*)d_in[10];
  const float* Wv2  = (const float*)d_in[11];  // (10240, 1024)
  const float* bv2  = (const float*)d_in[12];
  const float* lsc  = (const float*)d_in[13];  // scalar
  float* out = (float*)d_out;                  // (512, 100)

  // Workspace layout (~70 MB, fits 192 MB L2) with stream-ordered aliasing:
  //   R1 (21 MB): Wg bf16, then U f32 (Wg dead after GEMM1; U written in GEMM3)
  //   R2 (10.5 MB): xn bf16, then sim f32 (xn dead after GEMM1)
  char* ws = (char*)d_ws;
  size_t o_r1  = 0;                                  // 20,971,520 B
  size_t o_r2  = o_r1 + (size_t)VV * DD * DD * 2;    // 10,485,760 B
  size_t o_w2  = o_r2 + (size_t)BB * VV * DD * 2;    // Wv2 bf16: 20,971,520 B
  size_t o_y   = o_w2 + (size_t)VV * DD * DD * 2;    // y f32: 2 MB
  size_t o_g   = o_y  + (size_t)BB * DD * 4;         // g bf16: 1 MB
  size_t o_w1  = o_g  + (size_t)BB * DD * 2;         // Wv1 bf16: 2 MB
  size_t o_h   = o_w1 + (size_t)DD * DD * 2;         // h bf16: 1 MB
  size_t o_tx  = o_h  + (size_t)BB * DD * 2;         // txt bf16: 0.8 MB
  size_t o_ub  = o_tx + (size_t)CC * PP * DD * 2;    // img bf16: 10.5 MB
  size_t o_s1  = o_ub + (size_t)BB * VV * DD * 2;    // stats
  size_t o_s2  = o_s1 + 2 * (size_t)DD * 4;

  __bf16* Wgb  = (__bf16*)(ws + o_r1);
  float*  U    = (float*)(ws + o_r1);    // aliases Wgb after GEMM1
  __bf16* xnb  = (__bf16*)(ws + o_r2);
  float*  simb = (float*)(ws + o_r2);    // aliases xnb after GEMM1
  __bf16* Wv2b = (__bf16*)(ws + o_w2);
  float*  y    = (float*)(ws + o_y);
  __bf16* gb   = (__bf16*)(ws + o_g);
  __bf16* Wv1b = (__bf16*)(ws + o_w1);
  __bf16* hb   = (__bf16*)(ws + o_h);
  __bf16* txb  = (__bf16*)(ws + o_tx);
  __bf16* Ub   = (__bf16*)(ws + o_ub);
  float*  sc1  = (float*)(ws + o_s1);
  float*  sh1  = sc1 + DD;
  float*  sc2  = (float*)(ws + o_s2);
  float*  sh2  = sc2 + DD;

  // 0) One-time weight conversions to bf16.
  cvt_bf16_kernel<<<(VV * DD * DD) / 2048, 256, 0, stream>>>(Wg, Wgb, VV * DD * DD);
  cvt_bf16_kernel<<<(DD * DD) / 2048, 256, 0, stream>>>(Wv1, Wv1b, DD * DD);
  cvt_bf16_kernel<<<(VV * DD * DD) / 2048, 256, 0, stream>>>(Wv2, Wv2b, VV * DD * DD);

  // 1) BN1 stats over fusion-scaled feat (5120 rows) and apply -> bf16.
  bn_stats_kernel<<<DD, 256, 0, stream>>>(feat, fr, bn1g, bn1b, sc1, sh1,
                                          BB * VV, 1);
  bn_apply_kernel<<<(BB * VV * DD) / 256, 256, 0, stream>>>(
      feat, fr, sc1, sh1, xnb, BB * VV * DD, 1, 0);

  // 2) Text row-normalize -> bf16.
  row_norm_kernel<<<CC * PP, 256, 0, stream>>>(text, txb);

  // 3) GEMM1: y = xn(512,10240) @ Wg^T + bg -> f32 (512,1024)
  gemm_bf16_wmma<4, 1, false><<<dim3(DD / 64, BB / 256), 256, 0, stream>>>(
      xnb, VV * DD, Wgb, VV * DD, y, DD, bg, nullptr, VV * DD);

  // 4) BN2 stats + apply + ReLU -> bf16 g.
  bn_stats_kernel<<<DD, 256, 0, stream>>>(y, nullptr, bn2g, bn2b, sc2, sh2, BB, 0);
  bn_apply_kernel<<<(BB * DD) / 256, 256, 0, stream>>>(
      y, nullptr, sc2, sh2, gb, BB * DD, 0, 1);

  // 5) GEMM2: h = relu(g @ Wv1^T + bv1) -> bf16 (512,1024)
  gemm_bf16_wmma<4, 2, true><<<dim3(DD / 64, BB / 256), 256, 0, stream>>>(
      gb, DD, Wv1b, DD, hb, DD, bv1, nullptr, DD);

  // 6) GEMM3: U = relu(h @ Wv2^T + bv2)*0.6 + feat*0.4 -> f32 (512,10240)
  gemm_bf16_wmma<4, 3, false><<<dim3((VV * DD) / 64, BB / 256), 256, 0, stream>>>(
      hb, DD, Wv2b, DD, U, VV * DD, bv2, feat, DD);

  // 7) Row-normalize image features: U (5120,1024) f32 -> bf16 Ub.
  row_norm_kernel<<<BB * VV, 256, 0, stream>>>(U, Ub);

  // 8) Similarity GEMM: sim = Ub(5120,1024) @ txb(400,1024)^T -> f32 (5120,400)
  gemm_bf16_wmma<5, 0, false><<<dim3((CC * PP) / 80, (BB * VV) / 256), 256, 0, stream>>>(
      Ub, DD, txb, DD, simb, CC * PP, nullptr, nullptr, DD);

  // 9) Sinkhorn (10x4 per (b,c) problem) + logits.
  sinkhorn_logits_kernel<<<(BB * CC) / 256, 256, 0, stream>>>(simb, lsc, out);
}
